// ActMorphologyTransformer_32469952757982
// MI455X (gfx1250) — matrix operand order repaired and checked by reading, not compiled, not run on hardware
//
#include <hip/hip_runtime.h>
#include <math.h>

#define Bdim   16
#define Tdim   128
#define JMAXC  24
#define Hdim   256
#define HEADSC 8
#define DHC    32
#define Ldim   4
#define RANKC  8
#define Mnum   12
#define NROW   (Bdim * Tdim)      // 2048
#define NJ     (NROW * JMAXC)     // 49152

typedef __attribute__((ext_vector_type(16))) _Float16 v16h;
typedef __attribute__((ext_vector_type(8)))  _Float16 v8h;
typedef __attribute__((ext_vector_type(8)))  float    v8f;

__device__ __constant__ int d_num_global[Mnum] = {1, 0, 1, 1, 0, 1, 1, 1, 0, 1, 1, 1};

// ---------------------------------------------------------------------------
// Weight transpose + f32 -> f16 convert:  src (L, K, N) f32  ->  dst (L, N, K) f16
// ---------------------------------------------------------------------------
__global__ void transpose_f32_f16(const float* __restrict__ src, _Float16* __restrict__ dst,
                                  int K, int N) {
    int l = blockIdx.z;
    size_t per = (size_t)K * N;
    int idx = blockIdx.x * blockDim.x + threadIdx.x;
    if (idx >= K * N) return;
    int n = idx / K;
    int k = idx - n * K;
    dst[l * per + idx] = (_Float16)src[l * per + (size_t)k * N + n];
}

// ---------------------------------------------------------------------------
// Embedding: one block per (token, joint) row, thread = hidden index
// ---------------------------------------------------------------------------
__global__ __launch_bounds__(256)
void embed_kernel(const float* __restrict__ act,
                  const unsigned char* __restrict__ slide,
                  const unsigned char* __restrict__ hinge,
                  const unsigned char* __restrict__ glob,
                  const unsigned char* __restrict__ amask,
                  const int* __restrict__ m_idx,
                  const float* __restrict__ Ws, const float* __restrict__ bs,
                  const float* __restrict__ Wh, const float* __restrict__ bh,
                  const float* __restrict__ Wg, const float* __restrict__ Wact,
                  const float* __restrict__ pos,
                  float* __restrict__ x) {
    int rj = blockIdx.x;                 // n*J + j
    int h  = threadIdx.x;
    int n  = rj / JMAXC;
    int j  = rj - n * JMAXC;
    int b  = n / Tdim;
    int m  = m_idx[b];
    float a1 = act[rj];
    float e = 0.0f;
    if (slide[rj]) e = a1 * Ws[h] + bs[h];
    if (hinge[rj]) e = a1 * Wh[h] + bh[h];
    if (glob[rj] && d_num_global[m] > 0) e = a1 * Wg[m * Hdim + h];
    float y = e + (amask[rj] ? Wact[h] : 0.0f) + pos[((size_t)m * JMAXC + j) * Hdim + h];
    x[(size_t)rj * Hdim + h] = y;
}

// ---------------------------------------------------------------------------
// LayerNorm over H=256 (one block per row). Writes f16 (GEMM input) or f32 (final out).
// ---------------------------------------------------------------------------
__global__ __launch_bounds__(256)
void ln_kernel(const float* __restrict__ x, const float* __restrict__ sc,
               const float* __restrict__ bi, _Float16* out16, float* out32) {
    __shared__ float red[256];
    int r = blockIdx.x, tid = threadIdx.x;
    float z = x[(size_t)r * Hdim + tid];
    red[tid] = z; __syncthreads();
    for (int s = 128; s > 0; s >>= 1) { if (tid < s) red[tid] += red[tid + s]; __syncthreads(); }
    float mu = red[0] * (1.0f / Hdim);
    __syncthreads();
    float d = z - mu;
    red[tid] = d * d; __syncthreads();
    for (int s = 128; s > 0; s >>= 1) { if (tid < s) red[tid] += red[tid + s]; __syncthreads(); }
    float var = red[0] * (1.0f / Hdim);
    float o = d * rsqrtf(var + 1e-5f) * sc[tid] + bi[tid];
    if (out16) out16[(size_t)r * Hdim + tid] = (_Float16)o;
    else       out32[(size_t)r * Hdim + tid] = o;
}

// ---------------------------------------------------------------------------
// WMMA GEMM:  C(Mx N) = A(M x K, f16 row-major) * Bt(N x K, f16)^T, f32 accum.
// 128 threads = 4 waves, each wave a 16x64 strip. modes: 0 = +bias -> f16
// 1 = gelu(+bias) -> f16 ; 2 = resid += gvec*( +bias ) (f32, fused residual)
// ---------------------------------------------------------------------------
__global__ __launch_bounds__(128)
void gemm_wmma(const _Float16* __restrict__ A, const _Float16* __restrict__ Bt,
               const float* __restrict__ bias, const float* __restrict__ gvec,
               _Float16* __restrict__ out16, float* __restrict__ resid,
               int N, int K, int mode) {
    int lane = threadIdx.x & 31;
    int wave = threadIdx.x >> 5;
    int ml   = lane & 15;
    int g    = lane >> 4;
    int m0   = blockIdx.y * 64 + wave * 16;
    int n0   = blockIdx.x * 64;

    v8f acc[4] = {};
    const _Float16* arow = A + (size_t)(m0 + ml) * K;

    for (int k0 = 0; k0 < K; k0 += 32) {
        // A fragment: elem i -> K = k0 + i + 16g (contiguous, matches B)
        v16h a = *(const v16h*)(arow + k0 + 16 * g);
#pragma unroll
        for (int t = 0; t < 4; t++) {
            // B fragment: col = ml, elem i -> K = k0 + i + 16g (contiguous in Bt row)
            int n = n0 + t * 16 + ml;
            v16h bfr = *(const v16h*)(Bt + (size_t)n * K + k0 + 16 * g);
            acc[t] = __builtin_amdgcn_wmma_f32_16x16x32_f16(
                false, a, false, bfr, (short)0, acc[t], false, false);
        }
    }

#pragma unroll
    for (int t = 0; t < 4; t++) {
        int c = n0 + t * 16 + ml;
#pragma unroll
        for (int v = 0; v < 8; v++) {
            int r = m0 + v + 8 * g;                 // C layout: lanes16-31 -> M = v+8
            float val = acc[t][v] + bias[c];
            size_t idx = (size_t)r * N + c;
            if (mode == 0) {
                out16[idx] = (_Float16)val;
            } else if (mode == 1) {
                float u  = 0.7978845608028654f * (val + 0.044715f * val * val * val);
                float ge = 0.5f * val * (1.0f + tanhf(u));
                out16[idx] = (_Float16)ge;
            } else {
                resid[idx] += gvec[c] * val;
            }
        }
    }
}

// ---------------------------------------------------------------------------
// LoRA: qkv(r, 0:768) += (h16(r,:) @ lA[m]) @ lB[m]   (rank 8)
// ---------------------------------------------------------------------------
__global__ __launch_bounds__(256)
void lora_kernel(const _Float16* __restrict__ h16, const float* __restrict__ lA,
                 const float* __restrict__ lB, const int* __restrict__ m_idx,
                 _Float16* __restrict__ qkv) {
    __shared__ float red[256];
    __shared__ float tmp[RANKC];
    int r = blockIdx.x, tid = threadIdx.x;
    int b = r / (Tdim * JMAXC);
    int m = m_idx[b];
    const float* Am = lA + (size_t)m * Hdim * RANKC;
    const float* Bm = lB + (size_t)m * RANKC * 3 * Hdim;
    float hv = (float)h16[(size_t)r * Hdim + tid];
    for (int rk = 0; rk < RANKC; rk++) {
        red[tid] = hv * Am[tid * RANKC + rk];
        __syncthreads();
        for (int s = 128; s > 0; s >>= 1) { if (tid < s) red[tid] += red[tid + s]; __syncthreads(); }
        if (tid == 0) tmp[rk] = red[0];
        __syncthreads();
    }
    for (int c = tid; c < 3 * Hdim; c += 256) {
        float d = 0.0f;
#pragma unroll
        for (int rk = 0; rk < RANKC; rk++) d += tmp[rk] * Bm[rk * 3 * Hdim + c];
        size_t idx = (size_t)r * 3 * Hdim + c;
        qkv[idx] = (_Float16)((float)qkv[idx] + d);
    }
}

// ---------------------------------------------------------------------------
// Attention: one wave per (token, head). 24x24 masked softmax, DH=32.
// ---------------------------------------------------------------------------
__global__ __launch_bounds__(32)
void attn_kernel(const _Float16* __restrict__ qkv, const unsigned char* __restrict__ morph,
                 _Float16* __restrict__ o16) {
    __shared__ float ks[JMAXC][DHC];
    __shared__ float vs[JMAXC][DHC];
    int n  = blockIdx.x / HEADSC;
    int hd = blockIdx.x - n * HEADSC;
    int b  = n / Tdim;
    int j  = threadIdx.x;
    float q[DHC];
    if (j < JMAXC) {
        size_t base = ((size_t)n * JMAXC + j) * (3 * Hdim) + hd * DHC;
        for (int d = 0; d < DHC; d++) {
            q[d]     = (float)qkv[base + d];
            ks[j][d] = (float)qkv[base + Hdim + d];
            vs[j][d] = (float)qkv[base + 2 * Hdim + d];
        }
    }
    __syncthreads();
    if (j >= JMAXC) return;

    const float scale = 0.17677669529663687f; // 1/sqrt(32)
    float s[JMAXC];
    float mx = -3.402823466e38f;
    const unsigned char* mrow = morph + (size_t)b * JMAXC * JMAXC + (size_t)j * JMAXC;
    for (int jj = 0; jj < JMAXC; jj++) {
        float a = 0.0f;
        for (int d = 0; d < DHC; d++) a += q[d] * ks[jj][d];
        a *= scale;
        if (!mrow[jj]) a = -3.402823466e38f;
        s[jj] = a;
        mx = fmaxf(mx, a);
    }
    float sum = 0.0f;
    for (int jj = 0; jj < JMAXC; jj++) { s[jj] = __expf(s[jj] - mx); sum += s[jj]; }
    float inv = 1.0f / sum;
    size_t ob = ((size_t)n * JMAXC + j) * Hdim + hd * DHC;
    for (int d = 0; d < DHC; d++) {
        float a = 0.0f;
        for (int jj = 0; jj < JMAXC; jj++) a += s[jj] * vs[jj][d];
        o16[ob + d] = (_Float16)(a * inv);
    }
}

// ---------------------------------------------------------------------------
// Host orchestration
// ---------------------------------------------------------------------------
extern "C" void kernel_launch(void* const* d_in, const int* in_sizes, int n_in,
                              void* d_out, int out_size, void* d_ws, size_t ws_size,
                              hipStream_t stream) {
    const float*         act    = (const float*)d_in[0];
    const unsigned char* slide  = (const unsigned char*)d_in[1];
    const unsigned char* hinge  = (const unsigned char*)d_in[2];
    const unsigned char* glob   = (const unsigned char*)d_in[3];
    const unsigned char* amask  = (const unsigned char*)d_in[4];
    const unsigned char* morph  = (const unsigned char*)d_in[5];
    const int*           m_idx  = (const int*)d_in[6];
    const float* Ws    = (const float*)d_in[7];
    const float* bs    = (const float*)d_in[8];
    const float* Wh    = (const float*)d_in[9];
    const float* bh    = (const float*)d_in[10];
    const float* Wg    = (const float*)d_in[11];
    const float* Wact  = (const float*)d_in[12];
    const float* pos   = (const float*)d_in[13];
    const float* ln1_s = (const float*)d_in[14];
    const float* ln1_b = (const float*)d_in[15];
    const float* Wqkv  = (const float*)d_in[16];
    const float* bqkv  = (const float*)d_in[17];
    const float* loraA = (const float*)d_in[18];
    const float* loraB = (const float*)d_in[19];
    const float* Wo    = (const float*)d_in[20];
    const float* bo    = (const float*)d_in[21];
    const float* ln2_s = (const float*)d_in[22];
    const float* ln2_b = (const float*)d_in[23];
    const float* W1    = (const float*)d_in[24];
    const float* b1    = (const float*)d_in[25];
    const float* W2    = (const float*)d_in[26];
    const float* b2    = (const float*)d_in[27];
    const float* g1    = (const float*)d_in[28];
    const float* g2    = (const float*)d_in[29];
    const float* lnf_s = (const float*)d_in[30];
    const float* lnf_b = (const float*)d_in[31];

    // workspace layout
    char* p = (char*)d_ws;
    float*    x     = (float*)p;     p += (size_t)NJ * Hdim * 4;       // 50.3 MB
    _Float16* a16   = (_Float16*)p;  p += (size_t)NJ * Hdim * 2;       // 25.2 MB
    _Float16* o16   = (_Float16*)p;  p += (size_t)NJ * Hdim * 2;       // 25.2 MB
    _Float16* qkv16 = (_Float16*)p;  p += (size_t)NJ * 3 * Hdim * 2;   // 75.5 MB
    _Float16* mid16 = (_Float16*)p;  p += (size_t)NJ * 4 * Hdim * 2;   // 100.7 MB
    _Float16* wqkvT = (_Float16*)p;  p += (size_t)Ldim * 3 * Hdim * Hdim * 2;
    _Float16* woT   = (_Float16*)p;  p += (size_t)Ldim * Hdim * Hdim * 2;
    _Float16* w1T   = (_Float16*)p;  p += (size_t)Ldim * 4 * Hdim * Hdim * 2;
    _Float16* w2T   = (_Float16*)p;  p += (size_t)Ldim * 4 * Hdim * Hdim * 2;

    // weight transposes (f32 -> f16, N x K)
    transpose_f32_f16<<<dim3((Hdim * 3 * Hdim + 255) / 256, 1, Ldim), 256, 0, stream>>>(
        Wqkv, wqkvT, Hdim, 3 * Hdim);
    transpose_f32_f16<<<dim3((Hdim * Hdim + 255) / 256, 1, Ldim), 256, 0, stream>>>(
        Wo, woT, Hdim, Hdim);
    transpose_f32_f16<<<dim3((Hdim * 4 * Hdim + 255) / 256, 1, Ldim), 256, 0, stream>>>(
        W1, w1T, Hdim, 4 * Hdim);
    transpose_f32_f16<<<dim3((4 * Hdim * Hdim + 255) / 256, 1, Ldim), 256, 0, stream>>>(
        W2, w2T, 4 * Hdim, Hdim);

    // embedding -> x
    embed_kernel<<<NJ, 256, 0, stream>>>(act, slide, hinge, glob, amask, m_idx,
                                         Ws, bs, Wh, bh, Wg, Wact, pos, x);

    for (int l = 0; l < Ldim; l++) {
        // LN1 -> a16
        ln_kernel<<<NJ, 256, 0, stream>>>(x, ln1_s + l * Hdim, ln1_b + l * Hdim, a16, nullptr);
        // QKV GEMM (+bias) -> qkv16
        gemm_wmma<<<dim3((3 * Hdim) / 64, NJ / 64), 128, 0, stream>>>(
            a16, wqkvT + (size_t)l * 3 * Hdim * Hdim, bqkv + (size_t)l * 3 * Hdim,
            nullptr, qkv16, nullptr, 3 * Hdim, Hdim, 0);
        // LoRA update on qkv16
        lora_kernel<<<NJ, 256, 0, stream>>>(
            a16, loraA + (size_t)l * Mnum * Hdim * RANKC,
            loraB + (size_t)l * Mnum * RANKC * 3 * Hdim, m_idx, qkv16);
        // attention -> o16
        attn_kernel<<<NROW * HEADSC, 32, 0, stream>>>(qkv16, morph, o16);
        // O-proj GEMM, fused: x += g1 * (o @ Wo + bo)
        gemm_wmma<<<dim3(Hdim / 64, NJ / 64), 128, 0, stream>>>(
            o16, woT + (size_t)l * Hdim * Hdim, bo + (size_t)l * Hdim,
            g1 + (size_t)l * Hdim, nullptr, x, Hdim, Hdim, 2);
        // LN2 -> a16
        ln_kernel<<<NJ, 256, 0, stream>>>(x, ln2_s + l * Hdim, ln2_b + l * Hdim, a16, nullptr);
        // MLP1 GEMM, fused GELU -> mid16
        gemm_wmma<<<dim3((4 * Hdim) / 64, NJ / 64), 128, 0, stream>>>(
            a16, w1T + (size_t)l * 4 * Hdim * Hdim, b1 + (size_t)l * 4 * Hdim,
            nullptr, mid16, nullptr, 4 * Hdim, Hdim, 1);
        // MLP2 GEMM, fused: x += g2 * (... + b2)
        gemm_wmma<<<dim3(Hdim / 64, NJ / 64), 128, 0, stream>>>(
            mid16, w2T + (size_t)l * 4 * Hdim * Hdim, b2 + (size_t)l * Hdim,
            g2 + (size_t)l * Hdim, nullptr, x, Hdim, 4 * Hdim, 2);
    }

    // final LN -> d_out (f32)
    ln_kernel<<<NJ, 256, 0, stream>>>(x, lnf_s, lnf_b, nullptr, (float*)d_out);
}
